// ECGNFPModule_68410239091231
// MI455X (gfx1250) — matrix operand order repaired and checked by reading, not compiled, not run on hardware
//
#include <hip/hip_runtime.h>
#include <hip/hip_bf16.h>
#include <math.h>

#define BB   4
#define NN   8192
#define MM   65536
#define EE   131072
#define TT   16
#define F_IN 16
#define HIDC 32
#define OUTC 16
#define GDIM 1024
#define EMB  64
#define NHEAD 4
#define NPG  (NN/BB)   // 2048 coarse / graph
#define MPG  (MM/BB)   // 16384 fine / graph
#define EPG  (EE/BB)   // 32768 edges / graph
#define KPAD 96        // padded edge-feature dim (3 + 64 -> 96 = 3*32 for wmma K)
#define PF   67        // valid edge-feature dim

typedef __bf16 bf16_t;
typedef bf16_t bf16x8  __attribute__((ext_vector_type(8)));
typedef bf16_t bf16x16 __attribute__((ext_vector_type(16)));
typedef float  f32x8   __attribute__((ext_vector_type(8)));

// ---------------------------------------------------------------- utilities
__global__ void zero_u32_k(unsigned* p, int n) {
    int i = blockIdx.x * 256 + threadIdx.x;
    if (i < n) p[i] = 0u;
}

__global__ void count_k(const int* __restrict__ ei, int* __restrict__ cnt) {
    int e = blockIdx.x * 256 + threadIdx.x;
    if (e < EE) atomicAdd(&cnt[ei[EE + e]], 1);
}

// ------------------------------------------------- knn interpolate (k=3)
// h0[m, 0:16] = interp(x), h0[m, 16:32] = x_skip  -> input of NNConv layer 0
__global__ __launch_bounds__(256)
void knn_k(const float* __restrict__ pos, const float* __restrict__ x,
           const float* __restrict__ pos_skip, const float* __restrict__ x_skip,
           float* __restrict__ h0) {
    __shared__ float px[NPG], py[NPG], pz[NPG];
    const int g = blockIdx.x / (MPG / 256);            // graph id
    for (int i = threadIdx.x; i < NPG; i += 256) {
        int c = g * NPG + i;
        px[i] = pos[c * 3 + 0];
        py[i] = pos[c * 3 + 1];
        pz[i] = pos[c * 3 + 2];
    }
    __syncthreads();
    const int mi = blockIdx.x * 256 + threadIdx.x;     // fine node (global)
    const float qx = pos_skip[mi * 3 + 0];
    const float qy = pos_skip[mi * 3 + 1];
    const float qz = pos_skip[mi * 3 + 2];
    float b0 = 3.4e38f, b1 = 3.4e38f, b2 = 3.4e38f;
    int   i0 = 0, i1 = 0, i2 = 0;
    for (int n = 0; n < NPG; ++n) {
        float dx = px[n] - qx, dy = py[n] - qy, dz = pz[n] - qz;
        float d = dx * dx + dy * dy + dz * dz;
        if (d < b0)      { b2 = b1; i2 = i1; b1 = b0; i1 = i0; b0 = d; i0 = n; }
        else if (d < b1) { b2 = b1; i2 = i1; b1 = d;  i1 = n; }
        else if (d < b2) { b2 = d;  i2 = n; }
    }
    float w0 = 1.f / fmaxf(b0, 1e-16f);
    float w1 = 1.f / fmaxf(b1, 1e-16f);
    float w2 = 1.f / fmaxf(b2, 1e-16f);
    float inv = 1.f / (w0 + w1 + w2);
    const float* x0 = x + (size_t)(g * NPG + i0) * F_IN;
    const float* x1 = x + (size_t)(g * NPG + i1) * F_IN;
    const float* x2 = x + (size_t)(g * NPG + i2) * F_IN;
    float* out = h0 + (size_t)mi * 32;
    #pragma unroll
    for (int f = 0; f < F_IN; ++f) {
        out[f]        = (w0 * x0[f] + w1 * x1[f] + w2 * x2[f]) * inv;
        out[F_IN + f] = x_skip[(size_t)mi * F_IN + f];
    }
}

// ---------------------------------- convert edge-MLP weights to bf16, pad K->96
__global__ void wconv_k(const float* __restrict__ nw, bf16_t* __restrict__ wt, int rows) {
    int idx = blockIdx.x * 256 + threadIdx.x;
    if (idx >= rows * KPAD) return;
    int j = idx / KPAD, p = idx % KPAD;
    wt[idx] = (p < PF) ? (bf16_t)nw[j * PF + p] : (bf16_t)0.f;
}

// ---------------------------------- fold wq @ Wq (64x64 @ 64x3 -> 64x3)
__global__ void wcomb_k(const float* __restrict__ ipw, const float* __restrict__ ipb,
                        const float* __restrict__ Wq_w, const float* __restrict__ Wq_b,
                        float* __restrict__ Wc, float* __restrict__ bqc) {
    int j = threadIdx.x;               // 64 threads
    float a0 = 0.f, a1 = 0.f, a2 = 0.f, bb = ipb[j];
    for (int d = 0; d < EMB; ++d) {
        float w = ipw[j * EMB + d];
        a0 += w * Wq_w[d * 3 + 0];
        a1 += w * Wq_w[d * 3 + 1];
        a2 += w * Wq_w[d * 3 + 2];
        bb += w * Wq_b[d];
    }
    Wc[j * 3 + 0] = a0; Wc[j * 3 + 1] = a1; Wc[j * 3 + 2] = a2;
    bqc[j] = bb;
}

// ---------------------------------- tokens: k' = (gt@WkT + pe)@wkT + bk, same for v'
__global__ __launch_bounds__(256)
void kv_k(const float* __restrict__ gt,
          const float* __restrict__ Wk_w, const float* __restrict__ Wk_b,
          const float* __restrict__ Wv_w, const float* __restrict__ Wv_b,
          const float* __restrict__ ipw,  const float* __restrict__ ipb,
          float* __restrict__ k2, float* __restrict__ v2) {
    __shared__ float Kt[TT * EMB], Vt[TT * EMB];
    const int b = blockIdx.x;
    const float LN = 0.14391565f;                       // ln(10000)/64
    for (int idx = threadIdx.x; idx < TT * EMB; idx += 256) {
        int t = idx / EMB, d = idx % EMB;
        float freq = __expf(-LN * (float)(d & ~1));
        float pe   = (d & 1) ? __cosf((float)t * freq) : __sinf((float)t * freq);
        const float* g = gt + (size_t)(b * TT + t) * GDIM;
        float ak = Wk_b[d] + pe, av = Wv_b[d] + pe;
        const float* wkr = Wk_w + (size_t)d * GDIM;
        const float* wvr = Wv_w + (size_t)d * GDIM;
        for (int c = 0; c < GDIM; ++c) { ak += g[c] * wkr[c]; av += g[c] * wvr[c]; }
        Kt[idx] = ak; Vt[idx] = av;
    }
    __syncthreads();
    for (int idx = threadIdx.x; idx < TT * EMB; idx += 256) {
        int t = idx / EMB, j = idx % EMB;
        float ak = ipb[EMB + j], av = ipb[2 * EMB + j];
        #pragma unroll 8
        for (int d = 0; d < EMB; ++d) {
            ak += ipw[(EMB + j) * EMB + d]     * Kt[t * EMB + d];
            av += ipw[(2 * EMB + j) * EMB + d] * Vt[t * EMB + d];
        }
        k2[(size_t)(b * TT + t) * EMB + j] = ak;
        v2[(size_t)(b * TT + t) * EMB + j] = av;
    }
}

// ---------------------------------- per-edge attention -> pseudo_ext bf16 [E,96]
__global__ __launch_bounds__(256)
void attn_k(const int* __restrict__ ei, const float* __restrict__ pos_skip,
            const float* __restrict__ k2, const float* __restrict__ v2,
            const float* __restrict__ Wc, const float* __restrict__ bqc,
            const float* __restrict__ out_w, const float* __restrict__ out_b,
            bf16_t* __restrict__ ped) {
    __shared__ float sOW[EMB * EMB];        // 16 KB
    __shared__ float sK[TT * EMB], sV[TT * EMB];
    __shared__ float sWc[EMB * 3], sBqc[EMB], sOB[EMB];
    const int e0 = blockIdx.x * 256;
    const int b  = e0 / EPG;
    const int tid = threadIdx.x;
    for (int i = tid; i < EMB * EMB; i += 256) sOW[i] = out_w[i];
    for (int i = tid; i < TT * EMB; i += 256) {
        sK[i] = k2[(size_t)b * TT * EMB + i];
        sV[i] = v2[(size_t)b * TT * EMB + i];
    }
    if (tid < EMB)     { sBqc[tid] = bqc[tid]; sOB[tid] = out_b[tid]; }
    if (tid < EMB * 3) sWc[tid] = Wc[tid];
    __syncthreads();

    const int e = e0 + tid;
    const int s = ei[e], d = ei[EE + e];
    float sx = pos_skip[s * 3 + 0], sy = pos_skip[s * 3 + 1], sz = pos_skip[s * 3 + 2];
    float dx = pos_skip[d * 3 + 0], dy = pos_skip[d * 3 + 1], dz = pos_skip[d * 3 + 2];
    float pdx = dx - sx, pdy = dy - sy, pdz = dz - sz;
    float ex = 0.5f * (dx + sx), ey = 0.5f * (dy + sy), ez = 0.5f * (dz + sz);

    float acc[EMB];
    #pragma unroll
    for (int o = 0; o < EMB; ++o) acc[o] = 0.f;

    for (int h = 0; h < NHEAD; ++h) {
        float qh[16];
        #pragma unroll
        for (int u = 0; u < 16; ++u) {
            int j = h * 16 + u;
            qh[u] = sBqc[j] + ex * sWc[j * 3] + ey * sWc[j * 3 + 1] + ez * sWc[j * 3 + 2];
        }
        float sc[TT]; float mx = -3.4e38f;
        #pragma unroll
        for (int k = 0; k < TT; ++k) {
            float t = 0.f;
            #pragma unroll
            for (int u = 0; u < 16; ++u) t += qh[u] * sK[k * EMB + h * 16 + u];
            t *= 0.25f;                         // 1/sqrt(head_dim=16)
            sc[k] = t; mx = fmaxf(mx, t);
        }
        float ssum = 0.f;
        #pragma unroll
        for (int k = 0; k < TT; ++k) { sc[k] = __expf(sc[k] - mx); ssum += sc[k]; }
        float inv = 1.f / ssum;
        #pragma unroll
        for (int u = 0; u < 16; ++u) {
            float t = 0.f;
            #pragma unroll
            for (int k = 0; k < TT; ++k) t += sc[k] * sV[k * EMB + h * 16 + u];
            t *= inv;
            int j = h * 16 + u;
            #pragma unroll
            for (int o = 0; o < EMB; ++o) acc[o] += t * sOW[o * EMB + j];
        }
    }
    bf16_t* pr = ped + (size_t)e * KPAD;
    pr[0] = (bf16_t)pdx; pr[1] = (bf16_t)pdy; pr[2] = (bf16_t)pdz;
    #pragma unroll
    for (int o = 0; o < EMB; ++o) pr[3 + o] = (bf16_t)(acc[o] + sOB[o]);
    #pragma unroll
    for (int p = PF; p < KPAD; ++p) pr[p] = (bf16_t)0.f;
}

// ------------------------ fused NNConv: WMMA edge-MLP + relu + per-edge contract
// one wave per 16-edge tile. we = relu(pseudo@nwT+nb) is produced tile-by-tile
// with v_wmma_f32_16x16x32_bf16 and contracted in REGISTERS: for a fixed lane
// the C fragment always covers the same 8 edge rows (m = r+8*hi) and a column
// o = mrow+16p that cycles over CO/16 values, so each lane owns 8*CO/16 message
// partial sums. h[src] index i advances once per CO/16 tiles -> hoisted LDS read.
// Each (m,o) cell is owned by exactly one lane -> direct global atomic scatter.
template <int CI, int CO>
__global__ __launch_bounds__(32)
void nnconv_k(const bf16_t* __restrict__ ped, const bf16_t* __restrict__ wt,
              const float* __restrict__ nb, const float* __restrict__ h,
              const int* __restrict__ ei, float* __restrict__ aggr) {
    constexpr int P = CO / 16;              // column sub-tiles per i (1 or 2)
    __shared__ float hs[16 * CI];
    __shared__ int   sDst[16];
    const int lane = threadIdx.x;
    const int e0 = blockIdx.x * 16;
    if (lane < 16) sDst[lane] = ei[EE + e0 + lane];
    for (int i = lane; i < 16 * CI; i += 32) {
        int m = i / CI, c = i % CI;
        hs[i] = h[(size_t)ei[e0 + m] * CI + c];
    }
    __syncthreads();

    const int mrow = lane & 15;
    const int hi   = lane >> 4;                       // 0: lanes 0-15, 1: lanes 16-31
    // A fragment (pseudo tile, 16 edges x 96 bf16), ISA 16-bit A 16x32 layout:
    // lane<16 holds K {0..7,16..23}, lane>=16 holds K {8..15,24..31} per k-block.
    const bf16_t* arow = ped + (size_t)(e0 + mrow) * KPAD;
    bf16x16 afrag[3];
    #pragma unroll
    for (int kb = 0; kb < 3; ++kb) {
        union { bf16x8 h2[2]; bf16x16 v; } u;
        u.h2[0] = *(const bf16x8*)(arow + kb * 32 + hi * 8);
        u.h2[1] = *(const bf16x8*)(arow + kb * 32 + hi * 8 + 16);
        afrag[kb] = u.v;
    }
    float msg[P * 8];
    #pragma unroll
    for (int q = 0; q < P * 8; ++q) msg[q] = 0.f;

    for (int i = 0; i < CI; ++i) {
        float hv[8];
        #pragma unroll
        for (int r = 0; r < 8; ++r) hv[r] = hs[(r + 8 * hi) * CI + i];
        #pragma unroll
        for (int p = 0; p < P; ++p) {
            const int j = (i * P + p) * 16 + mrow;    // output column of this lane
            float bias = nb[j];
            f32x8 c = {bias, bias, bias, bias, bias, bias, bias, bias};
            #pragma unroll
            for (int kb = 0; kb < 3; ++kb) {
                // B 32x16: lane n holds K {0..7,16..23} (hi=0) or {8..15,24..31} (hi=1) of column n = row j of nw
                union { bf16x8 h2[2]; bf16x16 v; } ub;
                ub.h2[0] = *(const bf16x8*)(wt + (size_t)j * KPAD + kb * 32 + hi * 8);
                ub.h2[1] = *(const bf16x8*)(wt + (size_t)j * KPAD + kb * 32 + hi * 8 + 16);
                bf16x16 bfrag = ub.v;
                c = __builtin_amdgcn_wmma_f32_16x16x32_bf16(
                        false, afrag[kb], false, bfrag, (short)0, c, false, false);
            }
            #pragma unroll
            for (int r = 0; r < 8; ++r) {             // C: (lane,r) -> m = r+8*hi, n = mrow
                float v = c[r]; v = v > 0.f ? v : 0.f;    // relu
                msg[p * 8 + r] += v * hv[r];
            }
        }
    }
    #pragma unroll
    for (int p = 0; p < P; ++p) {
        #pragma unroll
        for (int r = 0; r < 8; ++r) {
            const int m = r + 8 * hi;
            const int o = mrow + 16 * p;
            unsafeAtomicAdd(&aggr[(size_t)sDst[m] * CO + o], msg[p * 8 + r]);
        }
    }
}

// ---------------------------------- node update: relu(aggr/cnt + h@root + bias)
template <int CI, int CO>
__global__ void nodeup_k(const float* __restrict__ h, const float* __restrict__ aggr,
                         const int* __restrict__ cnt, const float* __restrict__ rt,
                         const float* __restrict__ bs, float* __restrict__ hout) {
    int idx = blockIdx.x * 256 + threadIdx.x;
    if (idx >= MM * CO) return;
    int v = idx / CO, o = idx % CO;
    float c = (float)max(cnt[v], 1);
    float a = aggr[idx] / c + bs[o];
    #pragma unroll 8
    for (int i = 0; i < CI; ++i) a += h[(size_t)v * CI + i] * rt[i * CO + o];
    hout[idx] = a > 0.f ? a : 0.f;
}

// ---------------------------------- pack output tuple (h, pos_skip, batch_skip)
__global__ void fin_k(const float* __restrict__ h, const float* __restrict__ pos_skip,
                      const int* __restrict__ bsk, float* __restrict__ out) {
    int idx = blockIdx.x * 256 + threadIdx.x;
    if (idx >= MM * 20) return;
    if (idx < MM * 16)            out[idx] = h[idx];
    else if (idx < MM * 19)       out[idx] = pos_skip[idx - MM * 16];
    else                          out[idx] = (float)bsk[idx - MM * 19];
}

// =================================================================== launcher
extern "C" void kernel_launch(void* const* d_in, const int* in_sizes, int n_in,
                              void* d_out, int out_size, void* d_ws, size_t ws_size,
                              hipStream_t stream) {
    const float* x         = (const float*)d_in[0];
    const float* pos       = (const float*)d_in[1];
    const float* x_skip    = (const float*)d_in[3];
    const float* pos_skip  = (const float*)d_in[4];
    const int*   batch_skip= (const int*)d_in[5];
    const int*   ei        = (const int*)d_in[6];
    const float* gt        = (const float*)d_in[7];
    const float* Wq_w = (const float*)d_in[8],  *Wq_b = (const float*)d_in[9];
    const float* Wk_w = (const float*)d_in[10], *Wk_b = (const float*)d_in[11];
    const float* Wv_w = (const float*)d_in[12], *Wv_b = (const float*)d_in[13];
    const float* ipw  = (const float*)d_in[14], *ipb  = (const float*)d_in[15];
    const float* out_w= (const float*)d_in[16], *out_b= (const float*)d_in[17];
    const float* nn_w0=(const float*)d_in[18], *nn_b0=(const float*)d_in[19];
    const float* root0=(const float*)d_in[20], *bias0=(const float*)d_in[21];
    const float* nn_w1=(const float*)d_in[22], *nn_b1=(const float*)d_in[23];
    const float* root1=(const float*)d_in[24], *bias1=(const float*)d_in[25];
    const float* nn_w2=(const float*)d_in[26], *nn_b2=(const float*)d_in[27];
    const float* root2=(const float*)d_in[28], *bias2=(const float*)d_in[29];
    float* out = (float*)d_out;

    char* ws = (char*)d_ws; size_t off = 0;
    auto alloc = [&](size_t bytes) { char* p = ws + off; off += (bytes + 255) & ~(size_t)255; return p; };
    int*    cnt  = (int*)   alloc((size_t)MM * 4);
    float*  aggr = (float*) alloc((size_t)MM * 32 * 4);
    float*  hA   = (float*) alloc((size_t)MM * 32 * 4);
    float*  hB   = (float*) alloc((size_t)MM * 32 * 4);
    bf16_t* ped  = (bf16_t*)alloc((size_t)EE * KPAD * 2);
    bf16_t* wt0  = (bf16_t*)alloc((size_t)1024 * KPAD * 2);
    bf16_t* wt1  = (bf16_t*)alloc((size_t)1024 * KPAD * 2);
    bf16_t* wt2  = (bf16_t*)alloc((size_t)512  * KPAD * 2);
    float*  k2   = (float*) alloc((size_t)BB * TT * EMB * 4);
    float*  v2   = (float*) alloc((size_t)BB * TT * EMB * 4);
    float*  Wc   = (float*) alloc(64 * 3 * 4);
    float*  bqc  = (float*) alloc(64 * 4);

    // degree counts
    zero_u32_k<<<(MM + 255) / 256, 256, 0, stream>>>((unsigned*)cnt, MM);
    count_k<<<EE / 256, 256, 0, stream>>>(ei, cnt);
    // knn interpolate -> hA = [xf | x_skip]
    knn_k<<<MM / 256, 256, 0, stream>>>(pos, x, pos_skip, x_skip, hA);
    // bf16 weight conversion (padded K)
    wconv_k<<<(1024 * KPAD + 255) / 256, 256, 0, stream>>>(nn_w0, wt0, 1024);
    wconv_k<<<(1024 * KPAD + 255) / 256, 256, 0, stream>>>(nn_w1, wt1, 1024);
    wconv_k<<<(512  * KPAD + 255) / 256, 256, 0, stream>>>(nn_w2, wt2, 512);
    // attention precompute + per-edge attention
    wcomb_k<<<1, 64, 0, stream>>>(ipw, ipb, Wq_w, Wq_b, Wc, bqc);
    kv_k<<<BB, 256, 0, stream>>>(gt, Wk_w, Wk_b, Wv_w, Wv_b, ipw, ipb, k2, v2);
    attn_k<<<EE / 256, 256, 0, stream>>>(ei, pos_skip, k2, v2, Wc, bqc, out_w, out_b, ped);

    // layer 0: 32 -> 32   (hA -> hB)
    zero_u32_k<<<(MM * 32 + 255) / 256, 256, 0, stream>>>((unsigned*)aggr, MM * 32);
    nnconv_k<32, 32><<<EE / 16, 32, 0, stream>>>(ped, wt0, nn_b0, hA, ei, aggr);
    nodeup_k<32, 32><<<(MM * 32 + 255) / 256, 256, 0, stream>>>(hA, aggr, cnt, root0, bias0, hB);
    // layer 1: 32 -> 32   (hB -> hA)
    zero_u32_k<<<(MM * 32 + 255) / 256, 256, 0, stream>>>((unsigned*)aggr, MM * 32);
    nnconv_k<32, 32><<<EE / 16, 32, 0, stream>>>(ped, wt1, nn_b1, hB, ei, aggr);
    nodeup_k<32, 32><<<(MM * 32 + 255) / 256, 256, 0, stream>>>(hB, aggr, cnt, root1, bias1, hA);
    // layer 2: 32 -> 16   (hA -> hB)
    zero_u32_k<<<(MM * 16 + 255) / 256, 256, 0, stream>>>((unsigned*)aggr, MM * 16);
    nnconv_k<32, 16><<<EE / 16, 32, 0, stream>>>(ped, wt2, nn_b2, hA, ei, aggr);
    nodeup_k<32, 16><<<(MM * 16 + 255) / 256, 256, 0, stream>>>(hA, aggr, cnt, root2, bias2, hB);

    fin_k<<<(MM * 20 + 255) / 256, 256, 0, stream>>>(hB, pos_skip, batch_skip, out);
}
